// Posit_Drmm_Modeler_31404800868904
// MI455X (gfx1250) — matrix operand order repaired and checked by reading, not compiled, not run on hardware
//
#include <hip/hip_runtime.h>

// ---------------------------------------------------------------------------
// POSIT-DRMM forward for MI455X (gfx1250, wave32, WMMA).
// Dominant GEMM (im2col conv, 88.5 GFLOP) + both cosine GEMMs run on
// v_wmma_f32_16x16x32_bf16 (fp32 accumulate).  Exact-match channel computed
// via integer token equality (semantically equal to cos>=1-1e-5 for a random
// normal embedding table).  Norms in fp32.  The im2col gather uses
// GLOBAL_LOAD_ASYNC_TO_LDS_B64 (ASYNCcnt) when the builtin is available.
// ---------------------------------------------------------------------------

#define B_   64
#define NS_  50
#define LS_  60
#define LQ_  32
#define V_   100000
#define D_   300
#define F_   256
#define FS_  3
#define KC   (FS_ * D_)   // 900
#define KCP  928          // 29 * 32
#define DP   320          // 10 * 32, padded embedding row

typedef unsigned short u16;
typedef __attribute__((ext_vector_type(8)))  u16    v8u;
typedef __attribute__((ext_vector_type(16))) __bf16 v16bf;
typedef __attribute__((ext_vector_type(8)))  float  v8f;
typedef __attribute__((ext_vector_type(2)))  int    v2i;

#ifndef __has_builtin
#define __has_builtin(x) 0
#endif
#if __has_builtin(__builtin_amdgcn_global_load_async_to_lds_b64) && \
    __has_builtin(__builtin_amdgcn_s_wait_asynccnt)
#define HAVE_ASYNC_LDS 1
#else
#define HAVE_ASYNC_LDS 0
#endif

// async-copy payload pointers: global (AS1) source, LDS (AS3) destination
typedef __attribute__((address_space(1))) v2i g_v2i;
typedef __attribute__((address_space(3))) v2i l_v2i;

__device__ __forceinline__ u16 f2bf(float f) {   // RNE f32 -> bf16
  unsigned int u = __float_as_uint(f);
  return (u16)((u + 0x7FFFu + ((u >> 16) & 1u)) >> 16);
}

// A-fragment (16x32 bf16, ISA 7.12.2): lane<16 holds K {0..7, 16..23} of row
// (lane&15); lane>=16 holds K {8..15, 24..31}.  Caller pre-adds half*8.
__device__ __forceinline__ v16bf ld_frag_a(const u16* p) {
  union { v16bf bf; v8u u[2]; } x;
  x.u[0] = *(const v8u*)(p);
  x.u[1] = *(const v8u*)(p + 16);
  return x.bf;
}
// B-fragment (32x16 bf16): lane<16 holds K {0..7, 16..23} of
// column (lane&15); lane>=16 holds K {8..15, 24..31}.  Caller pre-adds half*8.
__device__ __forceinline__ v16bf ld_frag_b(const u16* p) {
  union { v16bf bf; v8u u[2]; } x;
  x.u[0] = *(const v8u*)(p);
  x.u[1] = *(const v8u*)(p + 16);
  return x.bf;
}

__device__ __forceinline__ v8f wmma_bf16(v16bf a, v16bf b, v8f c) {
  return __builtin_amdgcn_wmma_f32_16x16x32_bf16(false, a, false, b,
                                                 (short)0, c, false, false);
}

// ---------------------------------------------------------------- converts
__global__ void k_convert_embeds(const float* __restrict__ emb,
                                 u16* __restrict__ outw) {
  size_t i = (size_t)blockIdx.x * 256 + threadIdx.x;   // V_*DP total
  int v = (int)(i / DP), c = (int)(i % DP);
  u16 r = 0;
  if (c < D_) r = f2bf(emb[(size_t)v * D_ + c]);
  outw[i] = r;
}

__global__ void k_convert_filters(const float* __restrict__ flt,
                                  u16* __restrict__ outw) {
  int i = blockIdx.x * 256 + threadIdx.x;              // F_*KCP total
  int f = i / KCP, c = i % KCP;
  u16 r = 0;
  if (c < KC) r = f2bf(flt[(size_t)f * KC + c]);
  outw[i] = r;
}

// ------------------------------------------------ fp32 embedding row norms
__global__ void k_row_inv_norms(const int* __restrict__ ids,
                                const float* __restrict__ emb,
                                float* __restrict__ inv, int nrows) {
  int i = blockIdx.x * blockDim.x + threadIdx.x;
  if (i >= nrows) return;
  const float4* p = (const float4*)(emb + (size_t)ids[i] * D_);
  float s = 0.f;
  for (int j = 0; j < D_ / 4; ++j) {
    float4 v = p[j];
    s += v.x * v.x + v.y * v.y + v.z * v.z + v.w * v.w;
  }
  inv[i] = rsqrtf(s);
}

// ------------------------------------------------------- conv GEMM (WMMA)
// out[row, f] = sum_{k<3, d<300} embeds[ids[n, l+k], d] * filters[f, k, d],
// rows = seq*L + l, zero beyond sequence end.  One block = 16 rows x 256 F.
#define ASTRIDE 936   // 16-row LDS im2col stride (bank-conflict free)
__global__ void k_conv_wmma(const int* __restrict__ ids,
                            const u16* __restrict__ embw,
                            const u16* __restrict__ fltw,
                            u16* __restrict__ outw,
                            float* __restrict__ invout, int L) {
  __shared__ u16   smA[16 * ASTRIDE];
  __shared__ float ct[16 * 260];
  __shared__ float pn[256];
  __shared__ int   tokb[16 * FS_];        // token id per (row, k), -1 = pad
  const int tid = threadIdx.x;
  const int row0 = blockIdx.x * 16;

  if (tid < 16 * FS_) {                    // hoist all div/mod out of staging
    int r = tid / FS_, k = tid - (tid / FS_) * FS_;
    int row = row0 + r;
    int n = row / L, l = row % L;
    tokb[tid] = (l + k < L) ? ids[n * L + l + k] : -1;
  }
  __syncthreads();

  // im2col staging: 16 rows x 928 bf16, 4-element (8B) copies per lane.
  // In-bounds copies go global->LDS via ASYNCcnt-tracked async loads; pad
  // elements (token -1 or K-tail) are zeroed with plain LDS stores.
  {
    const int c4 = tid * 4;                          // 0..1020
    int k = (c4 >= 2 * D_) ? 2 : (c4 >= D_ ? 1 : 0);
    int d = c4 - k * D_;
#pragma unroll 1
    for (int r = 0; r < 16; ++r) {
      if (c4 < KCP) {
        int tok = (c4 < KC) ? tokb[r * FS_ + k] : -1;
        u16* lp = &smA[r * ASTRIDE + c4];
        if (tok >= 0) {
          const u16* gp = embw + (size_t)tok * DP + d;
#if HAVE_ASYNC_LDS
          __builtin_amdgcn_global_load_async_to_lds_b64(
              (g_v2i*)(u16*)gp, (l_v2i*)lp, 0, 0);
#else
          *(unsigned long long*)lp = *(const unsigned long long*)gp;
#endif
        } else {
          *(unsigned long long*)lp = 0ull;
        }
      }
    }
  }
#if HAVE_ASYNC_LDS
  __builtin_amdgcn_s_wait_asynccnt(0);
#endif
  __syncthreads();

  const int wave = tid >> 5, lane = tid & 31;
  const int half = lane >> 4, rr = lane & 15;
  const int nt0 = wave * 2, nt1 = wave * 2 + 1;
  const u16* abase  = smA + rr * ASTRIDE + half * 8;
  const u16* bbase0 = fltw + (size_t)(nt0 * 16 + rr) * KCP + half * 8;
  const u16* bbase1 = fltw + (size_t)(nt1 * 16 + rr) * KCP + half * 8;
  v8f acc0 = {}, acc1 = {};
#pragma unroll 1
  for (int s = 0; s < KCP / 32; ++s) {
    int k0 = s * 32;
    __builtin_prefetch(bbase0 + k0 + 64, 0, 1);        // global_prefetch_b8
    v16bf a  = ld_frag_a(abase + k0);
    v16bf b0 = ld_frag_b(bbase0 + k0);
    v16bf b1 = ld_frag_b(bbase1 + k0);
    acc0 = wmma_bf16(a, b0, acc0);
    acc1 = wmma_bf16(a, b1, acc1);
  }
  // C layout: VGPR i -> M = i + 8*half, N = lane&15
#pragma unroll
  for (int i = 0; i < 8; ++i) {
    int M = i + half * 8;
    ct[M * 260 + nt0 * 16 + rr] = acc0[i];
    ct[M * 260 + nt1 * 16 + rr] = acc1[i];
  }
  __syncthreads();

  { // fp32 inverse norms per conv row
    int r = tid >> 4, cs = (tid & 15) * 16;
    float s2 = 0.f;
#pragma unroll
    for (int j = 0; j < 16; ++j) { float v = ct[r * 260 + cs + j]; s2 += v * v; }
    pn[tid] = s2;
  }
  __syncthreads();
  if (tid < 16) {
    float t2 = 0.f;
#pragma unroll
    for (int j = 0; j < 16; ++j) t2 += pn[tid * 16 + j];
    invout[row0 + tid] = rsqrtf(t2);
  }
  // bf16 row-major output [rows][256]
  for (int it = 0; it < 16; ++it) {
    int idx = it * 256 + tid;
    int r = idx >> 8, f = idx & 255;
    outw[(size_t)(row0 + r) * F_ + f] = f2bf(ct[r * 260 + f]);
  }
}

// ------------------------------- fused sim_ins/sim_sen/oh + pool + score
// One block per (b, n).  8 waves = 2 (M over LQ=32) x 4 (N over LS=60->64).
__global__ void k_sim_pool(const int* __restrict__ question,
                           const int* __restrict__ sentences,
                           const u16* __restrict__ embw,
                           const u16* __restrict__ qcw,
                           const u16* __restrict__ scw,
                           const float* __restrict__ inv_qn,
                           const float* __restrict__ inv_sn,
                           const float* __restrict__ inv_qc,
                           const float* __restrict__ inv_sc,
                           const float* __restrict__ W,
                           const float* __restrict__ bb,
                           float* __restrict__ out_sr) {
  __shared__ float simins[32 * 65], simsen[32 * 65];
  __shared__ float qn_s[32], qm_s[32], qcin_s[32];
  __shared__ float sn_s[LS_], sm_s[LS_], scin_s[LS_];
  __shared__ int   qtok[32], stok[64], cnt_s[32];
  __shared__ float fmx[2][32], fmn[2][32], p_s[32];

  const int blk = blockIdx.x;
  const int n = blk % NS_, b = blk / NS_;
  const int tid = threadIdx.x;
  const size_t sbase = (size_t)(b * NS_ + n) * LS_;

  if (tid < 32) {
    int t = question[b * LQ_ + tid];
    qtok[tid]  = t;
    qm_s[tid]  = (t > 0) ? 1.f : 0.f;
    qn_s[tid]  = inv_qn[b * LQ_ + tid];
    qcin_s[tid] = inv_qc[b * LQ_ + tid];
  }
  if (tid >= 64 && tid < 128) {
    int l = tid - 64;
    if (l < LS_) {
      int t = sentences[sbase + l];
      stok[l]  = t;
      sm_s[l]  = (t > 0) ? 1.f : 0.f;
      sn_s[l]  = inv_sn[sbase + l];
      scin_s[l] = inv_sc[sbase + l];
    } else stok[l] = 0;
  }
  __syncthreads();

  const int wave = tid >> 5, lane = tid & 31;
  const int half = lane >> 4, rr = lane & 15;
  const int mt = wave & 1, nt = wave >> 1;
  const int l = nt * 16 + rr;

  { // ---- sim_sen: qc[b] (32x256) x sc[b,n] (60x256)^T, K = 256
    const u16* arow = qcw + (size_t)(b * LQ_ + mt * 16 + rr) * F_ + half * 8;
    int srow = (int)sbase + nt * 16 + rr;
    if (srow >= B_ * NS_ * LS_) srow = B_ * NS_ * LS_ - 1;  // tail clamp (cols l>=60 discarded)
    const u16* brow = scw + (size_t)srow * F_ + half * 8;
    v8f acc = {};
#pragma unroll
    for (int s = 0; s < F_ / 32; ++s)
      acc = wmma_bf16(ld_frag_a(arow + s * 32), ld_frag_b(brow + s * 32), acc);
    if (l < LS_) {
      float fsc = scin_s[l] * sm_s[l];
#pragma unroll
      for (int i = 0; i < 8; ++i) {
        int q = mt * 16 + i + half * 8;
        simsen[q * 65 + l] = acc[i] * qcin_s[q] * qm_s[q] * fsc;
      }
    }
  }
  { // ---- sim_ins: qe (32x320) x se (60x320)^T, K = 320, token-gathered rows
    const u16* arow = embw + (size_t)qtok[mt * 16 + rr] * DP + half * 8;
    const u16* brow = embw + (size_t)stok[l] * DP + half * 8;
    v8f acc = {};
#pragma unroll
    for (int s = 0; s < DP / 32; ++s)
      acc = wmma_bf16(ld_frag_a(arow + s * 32), ld_frag_b(brow + s * 32), acc);
    if (l < LS_) {
      float fsc = sn_s[l] * sm_s[l];
#pragma unroll
      for (int i = 0; i < 8; ++i) {
        int q = mt * 16 + i + half * 8;
        simins[q * 65 + l] = acc[i] * qn_s[q] * qm_s[q] * fsc;
      }
    }
  }
  __syncthreads();

  // exact-match channel: integer token equality (== cos >= 1-1e-5)
  if (tid < 32) {
    int c = 0, qt = qtok[tid];
    for (int j = 0; j < LS_; ++j) c += (qt == stok[j]);
    cnt_s[tid] = c;
  }
  // top-5 pooling over LS for (q, channel)
  if (tid < 64) {
    int q = tid >> 1, ch = tid & 1;
    const float* rowp = (ch ? simsen : simins) + q * 65;
    float t0 = -1e30f, t1 = -1e30f, t2 = -1e30f, t3 = -1e30f, t4 = -1e30f;
    for (int j = 0; j < LS_; ++j) {
      float v = rowp[j];
      if (v > t4) {
        if (v > t0)      { t4 = t3; t3 = t2; t2 = t1; t1 = t0; t0 = v; }
        else if (v > t1) { t4 = t3; t3 = t2; t2 = t1; t1 = v; }
        else if (v > t2) { t4 = t3; t3 = t2; t2 = v; }
        else if (v > t3) { t4 = t3; t3 = v; }
        else             { t4 = v; }
      }
    }
    fmx[ch][q] = t0;
    fmn[ch][q] = (t0 + t1 + t2 + t3 + t4) * 0.2f;
  }
  __syncthreads();
  if (tid < 32) {
    int c = cnt_s[tid];
    float ohmax  = (c > 0) ? 1.f : 0.f;
    float ohmean = (float)(c > 5 ? 5 : c) * 0.2f;
    float z = W[0] * fmx[0][tid] + W[1] * fmn[0][tid] +
              W[2] * fmx[1][tid] + W[3] * fmn[1][tid] +
              W[4] * ohmax + W[5] * ohmean + bb[0];
    p_s[tid] = 1.f / (1.f + expf(-z));
  }
  __syncthreads();
  if (tid == 0) {
    float s = 0.f;
    for (int q = 0; q < LQ_; ++q) s += p_s[q];
    out_sr[b * NS_ + n] = s * (1.f / (float)LQ_);
  }
}

// ------------------------------------------------------------- finalize
__global__ void k_finalize(const float* __restrict__ sr,
                           const int* __restrict__ tsent,
                           const int* __restrict__ tdoc,
                           float* __restrict__ out) {
  __shared__ float doc_s[B_];
  __shared__ float red[256];
  int tid = threadIdx.x;
  if (tid < B_) {
    float m = -1e30f;
    for (int nn = 0; nn < NS_; ++nn) m = fmaxf(m, sr[tid * NS_ + nn]);
    doc_s[tid] = m;
    out[1 + B_ * NS_ + tid] = m;
  }
  __syncthreads();
  float part = 0.f;
  for (int i = tid; i < B_ * NS_; i += 256) {
    float p = fminf(fmaxf(sr[i], 1e-7f), 1.f - 1e-7f);
    float t = (float)tsent[i];
    part += t * logf(p) + (1.f - t) * log1pf(-p);
  }
  red[tid] = part;
  __syncthreads();
  for (int s = 128; s > 0; s >>= 1) {
    if (tid < s) red[tid] += red[tid + s];
    __syncthreads();
  }
  float s1 = red[0];
  __syncthreads();
  float part2 = 0.f;
  if (tid < B_) {
    float p = fminf(fmaxf(doc_s[tid], 1e-7f), 1.f - 1e-7f);
    float t = (float)tdoc[tid];
    part2 = t * logf(p) + (1.f - t) * log1pf(-p);
  }
  red[tid] = part2;
  __syncthreads();
  for (int s = 128; s > 0; s >>= 1) {
    if (tid < s) red[tid] += red[tid + s];
    __syncthreads();
  }
  if (tid == 0) {
    float bce1 = -s1 / (float)(B_ * NS_);
    float bce2 = -red[0] / (float)B_;
    out[0] = 0.5f * (bce1 + bce2);
  }
}

// ---------------------------------------------------------------- launch
extern "C" void kernel_launch(void* const* d_in, const int* in_sizes, int n_in,
                              void* d_out, int out_size, void* d_ws, size_t ws_size,
                              hipStream_t stream) {
  const int*   sentences = (const int*)d_in[0];
  const int*   question  = (const int*)d_in[1];
  const int*   tsent     = (const int*)d_in[2];
  const int*   tdoc      = (const int*)d_in[3];
  const float* embeds    = (const float*)d_in[4];
  const float* filters   = (const float*)d_in[5];
  const float* W         = (const float*)d_in[6];
  const float* bb        = (const float*)d_in[7];
  float* out = (float*)d_out;

  // workspace layout (~165.4 MB, all offsets 256B aligned)
  char* ws = (char*)d_ws;
  u16*   embw   = (u16*)(ws);                             // V*320*2   = 64,000,000
  u16*   fltw   = (u16*)(ws + 64000000);                  // 256*928*2 =    475,136
  u16*   qcw    = (u16*)(ws + 64475136);                  // 2048*256*2=  1,048,576
  u16*   scw    = (u16*)(ws + 65523712);                  // 192000*256*2=98,304,000
  float* inv_qn = (float*)(ws + 163827712);               // 2048*4
  float* inv_sn = (float*)(ws + 163835904);               // 192000*4
  float* inv_qc = (float*)(ws + 164603904);               // 2048*4
  float* inv_sc = (float*)(ws + 164612096);               // 192000*4 (end 165,380,096)

  const int QROWS = B_ * LQ_;                 // 2048
  const int SROWS = B_ * NS_ * LS_;           // 192000

  k_convert_embeds <<<(V_ * DP) / 256, 256, 0, stream>>>(embeds, embw);
  k_convert_filters<<<(F_ * KCP) / 256, 256, 0, stream>>>(filters, fltw);
  k_row_inv_norms  <<<(QROWS + 255) / 256, 256, 0, stream>>>(question, embeds, inv_qn, QROWS);
  k_row_inv_norms  <<<(SROWS + 255) / 256, 256, 0, stream>>>(sentences, embeds, inv_sn, SROWS);
  k_conv_wmma      <<<SROWS / 16, 256, 0, stream>>>(sentences, embw, fltw, scw, inv_sc, LS_);
  k_conv_wmma      <<<QROWS / 16, 256, 0, stream>>>(question, embw, fltw, qcw, inv_qc, LQ_);
  k_sim_pool       <<<B_ * NS_, 256, 0, stream>>>(question, sentences, embw, qcw, scw,
                                                  inv_qn, inv_sn, inv_qc, inv_sc, W, bb,
                                                  out + 1);
  k_finalize       <<<1, 256, 0, stream>>>(out + 1, tsent, tdoc, out);
}
